// ToggleHeteroGNN_v3_90855738180231
// MI455X (gfx1250) — matrix operand order repaired and checked by reading, not compiled, not run on hardware
//
#include <hip/hip_runtime.h>
#include <hip/hip_bf16.h>

typedef __attribute__((ext_vector_type(16))) _Float16 v16h;
typedef __attribute__((ext_vector_type(8)))  _Float16 v8h;
typedef __attribute__((ext_vector_type(8)))  float    v8f;

#define HDIM 64
#define LN_EPS 1e-5f

// ---------- helpers ----------

union AFrag { v16h v; _Float16 h[16]; };

__device__ __forceinline__ void cvt8(const float4 a, const float4 b, _Float16* h) {
  h[0] = (_Float16)a.x; h[1] = (_Float16)a.y; h[2] = (_Float16)a.z; h[3] = (_Float16)a.w;
  h[4] = (_Float16)b.x; h[5] = (_Float16)b.y; h[6] = (_Float16)b.z; h[7] = (_Float16)b.w;
}

__device__ __forceinline__ unsigned enc_f32(float f) {
  unsigned u = __float_as_uint(f);
  return (u & 0x80000000u) ? ~u : (u | 0x80000000u);
}
__device__ __forceinline__ float dec_f32(unsigned u) {
  u = (u & 0x80000000u) ? (u & 0x7FFFFFFFu) : ~u;
  return __uint_as_float(u);
}

// ---------- weight packing: w[K,64] (row-major) -> wt[n*Kpad + k] f16, zero-padded ----------
__global__ void pack_wt_kernel(const float* __restrict__ w, _Float16* __restrict__ wt,
                               int K, int Kpad) {
  int t = blockIdx.x * blockDim.x + threadIdx.x;
  if (t >= 64 * Kpad) return;
  int n = t / Kpad, k = t % Kpad;
  wt[t] = (k < K) ? (_Float16)w[k * 64 + n] : (_Float16)0.f;
}

// ---------- generic small encoder: out = (relu?)(in[:, col0:col0+Kin] @ w + b) ----------
__global__ void encode_kernel(const float* __restrict__ in, int ld, int col0, int Kin,
                              const float* __restrict__ w, const float* __restrict__ b,
                              float* __restrict__ out, int M, int N, int do_relu) {
  long t = (long)blockIdx.x * blockDim.x + threadIdx.x;
  if (t >= (long)M * N) return;
  int m = (int)(t / N), n = (int)(t % N);
  float s = b[n];
  for (int k = 0; k < Kin; ++k)
    s += in[(size_t)m * ld + col0 + k] * w[k * N + n];
  if (do_relu) s = fmaxf(s, 0.f);
  out[(size_t)m * N + n] = s;
}

// ---------- aggregation ----------
__global__ void agg_init_kernel(float* __restrict__ mean, unsigned* __restrict__ mx,
                                unsigned* __restrict__ mn, float* __restrict__ cnt,
                                float* __restrict__ deg, int Mdst, int Nsrc) {
  long t = (long)blockIdx.x * blockDim.x + threadIdx.x;
  long total = (long)Mdst * HDIM;
  if (t < total) { mean[t] = 0.f; mx[t] = 0u; mn[t] = 0xFFFFFFFFu; }
  if (t < Mdst) cnt[t] = 0.f;
  if (t < Nsrc) deg[t] = 0.f;
}

__global__ void deg_kernel(const int* __restrict__ src, float* __restrict__ deg, int E) {
  int e = blockIdx.x * blockDim.x + threadIdx.x;
  if (e >= E) return;
  atomicAdd(&deg[src[e]], 1.0f);
}

__global__ void agg_scatter_kernel(const float* __restrict__ x,
                                   const int* __restrict__ src, const int* __restrict__ dst,
                                   const float* __restrict__ deg,
                                   float* __restrict__ mean, unsigned* __restrict__ mx,
                                   unsigned* __restrict__ mn, float* __restrict__ cnt, int E) {
  long t = (long)blockIdx.x * blockDim.x + threadIdx.x;
  int e = (int)(t >> 4);
  int c = ((int)t & 15) * 4;
  if (e >= E) return;
  int s = src[e], d = dst[e];
  float inv = 1.0f / fmaxf(deg[s], 1.0f);
  if (c == 0) atomicAdd(&cnt[d], 1.0f);
  float4 v = *(const float4*)(x + (size_t)s * HDIM + c);
  size_t o = (size_t)d * HDIM + c;
  atomicAdd(&mean[o + 0], v.x * inv);
  atomicAdd(&mean[o + 1], v.y * inv);
  atomicAdd(&mean[o + 2], v.z * inv);
  atomicAdd(&mean[o + 3], v.w * inv);
  atomicMax(&mx[o + 0], enc_f32(v.x)); atomicMin(&mn[o + 0], enc_f32(v.x));
  atomicMax(&mx[o + 1], enc_f32(v.y)); atomicMin(&mn[o + 1], enc_f32(v.y));
  atomicMax(&mx[o + 2], enc_f32(v.z)); atomicMin(&mn[o + 2], enc_f32(v.z));
  atomicMax(&mx[o + 3], enc_f32(v.w)); atomicMin(&mn[o + 3], enc_f32(v.w));
}

__global__ void agg_finalize_kernel(unsigned* __restrict__ mx, unsigned* __restrict__ mn,
                                    const float* __restrict__ cnt, int Mdst) {
  long t = (long)blockIdx.x * blockDim.x + threadIdx.x;
  if (t >= (long)Mdst * HDIM) return;
  int d = (int)(t >> 6);
  bool has = cnt[d] > 0.f;
  float vmax = has ? dec_f32(mx[t]) : 0.f;
  float vmin = has ? dec_f32(mn[t]) : 0.f;
  ((float*)mx)[t] = vmax;
  ((float*)mn)[t] = vmin;
}

// ---------- WMMA node update: x = relu(layernorm(x + [mean|max|min] @ W + bias)) ----------
// W packed as wt[n*192 + k] f16. One wave = 16 rows x 64 cols, 8 waves/block.
__global__ void __launch_bounds__(256)
gnn_update_kernel(float* __restrict__ x,
                  const float* __restrict__ meanb, const float* __restrict__ mxf,
                  const float* __restrict__ mnf,
                  const _Float16* __restrict__ wt, const float* __restrict__ bias,
                  const float* __restrict__ g, const float* __restrict__ be, int M) {
  __shared__ float lds[8][16][65];
  const int lane = threadIdx.x & 31;
  const int wave = threadIdx.x >> 5;
  const int lg   = lane >> 4;   // lane group (0/1)
  const int li   = lane & 15;
  const long m0  = (long)blockIdx.x * 128 + wave * 16;

  if (m0 < M) {                 // uniform per wave: EXEC all-ones inside
    const long mrow = m0 + li;  // A-matrix row for this lane
    const float* bufs[3] = { meanb, mxf, mnf };
    v8f acc[4] = {v8f{}, v8f{}, v8f{}, v8f{}};
#pragma unroll
    for (int ks = 0; ks < 6; ++ks) {
      const int kb = ks * 32;
      AFrag af;
      {
        int k0 = kb + lg * 16;
        const float* p = bufs[k0 >> 6] + mrow * HDIM + (k0 & 63);
        cvt8(*(const float4*)p, *(const float4*)(p + 4), af.h);
      }
      {
        int k1 = kb + lg * 16 + 8;
        const float* p = bufs[k1 >> 6] + mrow * HDIM + (k1 & 63);
        cvt8(*(const float4*)p, *(const float4*)(p + 4), af.h + 8);
      }
#pragma unroll
      for (int nb = 0; nb < 4; ++nb) {
        v16h bv = *(const v16h*)(wt + (size_t)(nb * 16 + li) * 192 + kb + lg * 16);
        acc[nb] = __builtin_amdgcn_wmma_f32_16x16x32_f16(
            false, af.v, false, bv, (short)0, acc[nb], false, false);
      }
    }
#pragma unroll
    for (int nb = 0; nb < 4; ++nb) {
      int cc = nb * 16 + li;
      float bz = bias[cc];
#pragma unroll
      for (int r = 0; r < 8; ++r)
        lds[wave][r + lg * 8][cc] = acc[nb][r] + bz;
    }
  }
  __syncthreads();
  if (threadIdx.x < 128) {
    long m = (long)blockIdx.x * 128 + threadIdx.x;
    if (m < M) {
      float* row = &lds[threadIdx.x >> 4][threadIdx.x & 15][0];
      float* xr = x + m * HDIM;
      float mu = 0.f;
#pragma unroll
      for (int j = 0; j < HDIM; ++j) { float v = row[j] + xr[j]; row[j] = v; mu += v; }
      mu *= (1.f / HDIM);
      float var = 0.f;
#pragma unroll
      for (int j = 0; j < HDIM; ++j) { float d = row[j] - mu; var += d * d; }
      var *= (1.f / HDIM);
      float rs = rsqrtf(var + LN_EPS);
#pragma unroll
      for (int j = 0; j < HDIM; ++j) {
        float o = (row[j] - mu) * rs * g[j] + be[j];
        xr[j] = fmaxf(o, 0.f);
      }
    }
  }
}

// ---------- cell MLP stage 1: hidden = f16(relu([cell|emb|0] @ W1 + b1)), K=72 padded to 96 ----------
__global__ void __launch_bounds__(256)
mlp1_kernel(const float* __restrict__ cell, const float* __restrict__ emb,
            const _Float16* __restrict__ wt, const float* __restrict__ b1,
            _Float16* __restrict__ hidden, int M) {
  const int lane = threadIdx.x & 31;
  const int wave = threadIdx.x >> 5;
  const int lg = lane >> 4, li = lane & 15;
  const long m0 = (long)blockIdx.x * 128 + wave * 16;
  if (m0 >= M) return;
  const long mrow = m0 + li;
  v8f acc[4] = {v8f{}, v8f{}, v8f{}, v8f{}};
#pragma unroll
  for (int ks = 0; ks < 3; ++ks) {
    const int kb = ks * 32;
    AFrag af;
#pragma unroll
    for (int half = 0; half < 2; ++half) {
      int k = kb + lg * 16 + half * 8;
      _Float16* hh = af.h + half * 8;
      if (k < 64) {
        const float* p = cell + mrow * HDIM + k;
        cvt8(*(const float4*)p, *(const float4*)(p + 4), hh);
      } else if (k < 72) {   // k==64: exactly the 8-wide emb row
        const float* p = emb + mrow * 8;
        cvt8(*(const float4*)p, *(const float4*)(p + 4), hh);
      } else {
#pragma unroll
        for (int i = 0; i < 8; ++i) hh[i] = (_Float16)0.f;
      }
    }
#pragma unroll
    for (int nb = 0; nb < 4; ++nb) {
      v16h bv = *(const v16h*)(wt + (size_t)(nb * 16 + li) * 96 + kb + lg * 16);
      acc[nb] = __builtin_amdgcn_wmma_f32_16x16x32_f16(
          false, af.v, false, bv, (short)0, acc[nb], false, false);
    }
  }
#pragma unroll
  for (int nb = 0; nb < 4; ++nb) {
    int cc = nb * 16 + li;
    float bz = b1[cc];
#pragma unroll
    for (int r = 0; r < 8; ++r) {
      long m = m0 + r + lg * 8;
      hidden[m * HDIM + cc] = (_Float16)fmaxf(acc[nb][r] + bz, 0.f);
    }
  }
}

// ---------- cell MLP stage 2: cell_out = hidden @ W2 + b2 (f32 out, no relu) ----------
__global__ void __launch_bounds__(256)
mlp2_kernel(const _Float16* __restrict__ hidden, const _Float16* __restrict__ wt,
            const float* __restrict__ b2, float* __restrict__ cell_out, int M) {
  const int lane = threadIdx.x & 31;
  const int wave = threadIdx.x >> 5;
  const int lg = lane >> 4, li = lane & 15;
  const long m0 = (long)blockIdx.x * 128 + wave * 16;
  if (m0 >= M) return;
  const long mrow = m0 + li;
  v8f acc[4] = {v8f{}, v8f{}, v8f{}, v8f{}};
#pragma unroll
  for (int ks = 0; ks < 2; ++ks) {
    const int kb = ks * 32;
    AFrag af;
    *(v8h*)(af.h)     = *(const v8h*)(hidden + mrow * HDIM + kb + lg * 16);
    *(v8h*)(af.h + 8) = *(const v8h*)(hidden + mrow * HDIM + kb + lg * 16 + 8);
#pragma unroll
    for (int nb = 0; nb < 4; ++nb) {
      v16h bv = *(const v16h*)(wt + (size_t)(nb * 16 + li) * 64 + kb + lg * 16);
      acc[nb] = __builtin_amdgcn_wmma_f32_16x16x32_f16(
          false, af.v, false, bv, (short)0, acc[nb], false, false);
    }
  }
#pragma unroll
  for (int nb = 0; nb < 4; ++nb) {
    int cc = nb * 16 + li;
    float bz = b2[cc];
#pragma unroll
    for (int r = 0; r < 8; ++r) {
      long m = m0 + r + lg * 8;
      cell_out[m * HDIM + cc] = acc[nb][r] + bz;
    }
  }
}

// ---------- final projection: out = net @ w_out + b_out ----------
__global__ void out_kernel(const float* __restrict__ net, const float* __restrict__ w,
                           const float* __restrict__ b, float* __restrict__ out, int M) {
  int m = blockIdx.x * blockDim.x + threadIdx.x;
  if (m >= M) return;
  float s = b[0];
  const float* r = net + (size_t)m * HDIM;
#pragma unroll
  for (int j = 0; j < HDIM; ++j) s += r[j] * w[j];
  out[m] = s;
}

// ---------- host orchestration ----------
extern "C" void kernel_launch(void* const* d_in, const int* in_sizes, int n_in,
                              void* d_out, int out_size, void* d_ws, size_t ws_size,
                              hipStream_t stream) {
  const float* net_x    = (const float*)d_in[0];
  const float* pin_in_x = (const float*)d_in[1];
  const float* pin_out_x= (const float*)d_in[2];
  const float* cell_x   = (const float*)d_in[3];
  const int* src1 = (const int*)d_in[4];  const int* dst1 = (const int*)d_in[5];
  const int* src2 = (const int*)d_in[6];  const int* dst2 = (const int*)d_in[7];
  const int* src3 = (const int*)d_in[8];  const int* dst3 = (const int*)d_in[9];
  const int* src4 = (const int*)d_in[10]; const int* dst4 = (const int*)d_in[11];
  const float* w_ct = (const float*)d_in[12];  const float* b_ct = (const float*)d_in[13];
  const float* w_net= (const float*)d_in[14];  const float* b_net= (const float*)d_in[15];
  const float* w_pi = (const float*)d_in[16];  const float* b_pi = (const float*)d_in[17];
  const float* w_po = (const float*)d_in[18];  const float* b_po = (const float*)d_in[19];
  const float* w_cs = (const float*)d_in[20];  const float* b_cs = (const float*)d_in[21];
  const float* g_net = (const float*)d_in[22]; const float* be_net = (const float*)d_in[23];
  const float* g_pi  = (const float*)d_in[24]; const float* be_pi  = (const float*)d_in[25];
  const float* g_po  = (const float*)d_in[26]; const float* be_po  = (const float*)d_in[27];
  const float* g_cell= (const float*)d_in[28]; const float* be_cell= (const float*)d_in[29];
  const float* w_net_up = (const float*)d_in[30]; const float* b_net_up = (const float*)d_in[31];
  const float* w_pi_up  = (const float*)d_in[32]; const float* b_pi_up  = (const float*)d_in[33];
  const float* w_po_up  = (const float*)d_in[34]; const float* b_po_up  = (const float*)d_in[35];
  const float* w_cell_up= (const float*)d_in[36]; const float* b_cell_up= (const float*)d_in[37];
  const float* w_mlp1 = (const float*)d_in[38]; const float* b_mlp1 = (const float*)d_in[39];
  const float* w_mlp2 = (const float*)d_in[40]; const float* b_mlp2 = (const float*)d_in[41];
  const float* w_out  = (const float*)d_in[42]; const float* b_out  = (const float*)d_in[43];

  const int nNet  = in_sizes[0] / 4;
  const int nPi   = in_sizes[1] / 3;
  const int nPo   = in_sizes[2] / 3;
  const int nCell = in_sizes[3] / 32;
  const int E1 = in_sizes[4], E2 = in_sizes[6], E3 = in_sizes[8], E4 = in_sizes[10];
  const int maxDst = nPi;  // largest destination set

  size_t off = 0;
  auto carve = [&](size_t bytes) -> void* {
    void* p = (char*)d_ws + off;
    off += (bytes + 255) & ~(size_t)255;
    return p;
  };
  float* netb    = (float*)carve((size_t)nNet  * HDIM * 4);
  float* pib     = (float*)carve((size_t)nPi   * HDIM * 4);
  float* pob     = (float*)carve((size_t)nPo   * HDIM * 4);
  float* cellb   = (float*)carve((size_t)nCell * HDIM * 4);
  float* embb    = (float*)carve((size_t)nCell * 8 * 4);
  float* cellout = (float*)carve((size_t)nCell * HDIM * 4);
  _Float16* hiddenb = (_Float16*)carve((size_t)nCell * HDIM * 2);
  float*    meanb = (float*)carve((size_t)maxDst * HDIM * 4);
  unsigned* mxb   = (unsigned*)carve((size_t)maxDst * HDIM * 4);
  unsigned* mnb   = (unsigned*)carve((size_t)maxDst * HDIM * 4);
  float* cntb = (float*)carve((size_t)maxDst * 4);
  float* degb = (float*)carve((size_t)maxDst * 4);
  _Float16* wt_pi   = (_Float16*)carve(64 * 192 * 2);
  _Float16* wt_cell = (_Float16*)carve(64 * 192 * 2);
  _Float16* wt_po   = (_Float16*)carve(64 * 192 * 2);
  _Float16* wt_net  = (_Float16*)carve(64 * 192 * 2);
  _Float16* wt_m1   = (_Float16*)carve(64 * 96 * 2);
  _Float16* wt_m2   = (_Float16*)carve(64 * 64 * 2);

  const int B = 256;
  auto nb = [](long n, int b) { return (int)((n + b - 1) / b); };

  // pack weights to f16 [64, Kpad]
  pack_wt_kernel<<<nb(64 * 192, B), B, 0, stream>>>(w_pi_up,   wt_pi,   192, 192);
  pack_wt_kernel<<<nb(64 * 192, B), B, 0, stream>>>(w_cell_up, wt_cell, 192, 192);
  pack_wt_kernel<<<nb(64 * 192, B), B, 0, stream>>>(w_po_up,   wt_po,   192, 192);
  pack_wt_kernel<<<nb(64 * 192, B), B, 0, stream>>>(w_net_up,  wt_net,  192, 192);
  pack_wt_kernel<<<nb(64 * 96,  B), B, 0, stream>>>(w_mlp1, wt_m1, 72, 96);
  pack_wt_kernel<<<nb(64 * 64,  B), B, 0, stream>>>(w_mlp2, wt_m2, 64, 64);

  // encoders
  encode_kernel<<<nb((long)nNet  * 64, B), B, 0, stream>>>(net_x,    4, 0, 4,  w_net, b_net, netb,  nNet,  64, 1);
  encode_kernel<<<nb((long)nPi   * 64, B), B, 0, stream>>>(pin_in_x, 3, 0, 3,  w_pi,  b_pi,  pib,   nPi,   64, 1);
  encode_kernel<<<nb((long)nPo   * 64, B), B, 0, stream>>>(pin_out_x,3, 0, 3,  w_po,  b_po,  pob,   nPo,   64, 1);
  encode_kernel<<<nb((long)nCell * 64, B), B, 0, stream>>>(cell_x,  32, 26, 6, w_cs,  b_cs,  cellb, nCell, 64, 1);
  encode_kernel<<<nb((long)nCell * 8,  B), B, 0, stream>>>(cell_x,  32, 0, 26, w_ct,  b_ct,  embb,  nCell, 8,  0);

  auto aggregate = [&](const float* srcx, const int* src, const int* dst,
                       int nSrc, int nDst, int E) {
    agg_init_kernel<<<nb((long)nDst * 64, B), B, 0, stream>>>(meanb, mxb, mnb, cntb, degb, nDst, nSrc);
    deg_kernel<<<nb(E, B), B, 0, stream>>>(src, degb, E);
    agg_scatter_kernel<<<nb((long)E * 16, B), B, 0, stream>>>(srcx, src, dst, degb, meanb, mxb, mnb, cntb, E);
    agg_finalize_kernel<<<nb((long)nDst * 64, B), B, 0, stream>>>(mxb, mnb, cntb, nDst);
  };

  for (int layer = 0; layer < 4; ++layer) {
    // net -> pin_in
    aggregate(netb, src1, dst1, nNet, nPi, E1);
    gnn_update_kernel<<<nb(nPi, 128), 256, 0, stream>>>(
        pib, meanb, (const float*)mxb, (const float*)mnb, wt_pi, b_pi_up, g_pi, be_pi, nPi);
    // pin_in -> cell
    aggregate(pib, src2, dst2, nPi, nCell, E2);
    gnn_update_kernel<<<nb(nCell, 128), 256, 0, stream>>>(
        cellb, meanb, (const float*)mxb, (const float*)mnb, wt_cell, b_cell_up, g_cell, be_cell, nCell);
    // cell MLP
    mlp1_kernel<<<nb(nCell, 128), 256, 0, stream>>>(cellb, embb, wt_m1, b_mlp1, hiddenb, nCell);
    mlp2_kernel<<<nb(nCell, 128), 256, 0, stream>>>(hiddenb, wt_m2, b_mlp2, cellout, nCell);
    // cell_out -> pin_out
    aggregate(cellout, src3, dst3, nCell, nPo, E3);
    gnn_update_kernel<<<nb(nPo, 128), 256, 0, stream>>>(
        pob, meanb, (const float*)mxb, (const float*)mnb, wt_po, b_po_up, g_po, be_po, nPo);
    // pin_out -> net
    aggregate(pob, src4, dst4, nPo, nNet, E4);
    gnn_update_kernel<<<nb(nNet, 128), 256, 0, stream>>>(
        netb, meanb, (const float*)mxb, (const float*)mnb, wt_net, b_net_up, g_net, be_net, nNet);
  }

  out_kernel<<<nb(nNet, B), B, 0, stream>>>(netb, w_out, b_out, (float*)d_out, nNet);
}
